// CausalSelfAttention_39410619908235
// MI455X (gfx1250) — compile-verified, matrix operand order checked by reading, not run on hardware
//
#include <hip/hip_runtime.h>
#include <hip/hip_bf16.h>
#include <stdint.h>

// ---------------------------------------------------------------------------
// Types for CDNA5 WMMA (wave32): v_wmma_f32_16x16x32_bf16
// ---------------------------------------------------------------------------
typedef __bf16 bf16;
typedef __attribute__((ext_vector_type(16))) __bf16 v16bf;
typedef __attribute__((ext_vector_type(8)))  __bf16 v8bf;
typedef __attribute__((ext_vector_type(8)))  float  v8f;
typedef uint32_t u32;
typedef uint64_t u64;
typedef __attribute__((ext_vector_type(4))) u32 u32x4;
typedef __attribute__((ext_vector_type(8))) u32 u32x8;

union BFrag { v16bf v; v8bf h[2]; };

#define WMMA_BF16(a, b, c) \
  __builtin_amdgcn_wmma_f32_16x16x32_bf16(false, (a), false, (b), (short)0, (c), false, false)

// fp32 -> bf16 round-to-nearest-even, no reliance on __bf16 cast codegen
static __device__ __forceinline__ bf16 f2bf(float f) {
  union { float f; uint32_t u; } in; in.f = f;
  uint32_t r = (in.u + 0x7FFFu + ((in.u >> 16) & 1u)) >> 16;
  union { uint16_t u; bf16 b; } out; out.u = (uint16_t)r;
  return out.b;
}

// Load one 16x32 (A) / 32x16 (B) bf16 fragment for lane-half h.
// Chunk layout per ISA 7.12.2: VGPR0-3 hold K = 8h..8h+7, VGPR4-7 hold K = 16+8h..
static __device__ __forceinline__ v16bf load_frag(const bf16* base, int h) {
  BFrag f;
  f.h[0] = *(const v8bf*)(base + 8 * h);
  f.h[1] = *(const v8bf*)(base + 16 + 8 * h);
  return f.v;
}

// Generic LDS pointer -> 32-bit LDS byte offset (aperture low bits are the offset)
static __device__ __forceinline__ u32 lds_byte_offset(const void* p) {
  return (u32)(uintptr_t)p;
}

// ---------------------------------------------------------------------------
// Tensor Data Mover: 2D tile Global -> LDS (inline asm, portable per bridge doc).
// Descriptor per CDNA5 ISA ch.8 (CDNA5 ISA).
//   group0: [0]=count=1, [1]=lds_addr, [2]=ga[31:0], [3]=ga[56:32] | type(2)<<30
//   group1: data_size/pad fields, tensor_dim0/1, tile_dim0/1, dim0 stride
// All inputs are wave-uniform -> SGPR operands. Completion via TENSORcnt.
// ---------------------------------------------------------------------------
static __device__ __forceinline__ void tdm_load_2d(
    u32 lds_addr, const void* gptr,
    u32 g1w0, u32 g1w1, u32 g1w2, u32 g1w3, u32 g1w4, u32 g1w5) {
  u64 ga = (u64)(uintptr_t)gptr;
  u32x4 g0;
  g0[0] = 1u;                                              // count=1, user mode
  g0[1] = lds_addr;                                        // LDS byte address
  g0[2] = (u32)ga;                                         // global addr lo
  g0[3] = ((u32)(ga >> 32) & 0x01FFFFFFu) | (2u << 30);    // addr hi | type=2
  u32x8 g1;
  g1[0] = g1w0; g1[1] = g1w1; g1[2] = g1w2; g1[3] = g1w3;
  g1[4] = g1w4; g1[5] = g1w5; g1[6] = 0u;   g1[7] = 0u;
  asm volatile("tensor_load_to_lds %0, %1" :: "s"(g0), "s"(g1) : "memory");
}

// ---------------------------------------------------------------------------
// Problem constants
// ---------------------------------------------------------------------------
#define Bb 2
#define Tt 2048
#define Cc 1024
#define Hh 16
#define HD 64

// ---------------------------------------------------------------------------
// Kernel 1: fp32 -> bf16 conversion (x4 vectorized)
// ---------------------------------------------------------------------------
__global__ void cvt_f32_to_bf16(const float* __restrict__ src,
                                bf16* __restrict__ dst, int n4) {
  int i = blockIdx.x * blockDim.x + threadIdx.x;
  if (i < n4) {
    float4 v = ((const float4*)src)[i];
    bf16* d = dst + (size_t)i * 4;
    d[0] = f2bf(v.x); d[1] = f2bf(v.y); d[2] = f2bf(v.z); d[3] = f2bf(v.w);
  }
}

// ---------------------------------------------------------------------------
// Kernel 2: tiled bf16 GEMM, C(f32) = A(bf16,[M,K]) x B(bf16,[K,N]).
// Compile-time dims -> address math folds into instruction offsets.
// Block 128x128, BK=32, 8 waves; each wave: 32x64 tile = 2x4 WMMA accumulators.
// Double-buffered pipeline: while WMMAs run on buffer `cur`, the Tensor Data
// Mover streams the next A tile into `nxt` (TENSORcnt waited only after
// compute), and the next B tile's global loads issue before the WMMAs so
// their latency hides under the matrix ops. D# padding reproduces the 80B
// LDS row pitch (64B row + 4-DWORD pad every 16 DWORDs).
// ---------------------------------------------------------------------------
#define GBM 128
#define GBN 128
#define GBK 32
#define LDT 40  // pitch (elements): 80B rows -> 16B aligned chunks, non-pow2 banks

template <int M, int N, int K>
__global__ __launch_bounds__(256) void gemm_bf16_wmma(
    const bf16* __restrict__ A, const bf16* __restrict__ B,
    float* __restrict__ C) {
  __shared__ __align__(16) bf16 As[2][GBM * LDT];
  __shared__ __align__(16) bf16 Bs[2][GBN * LDT];  // stored N-major (transposed)

  const int tid  = threadIdx.x;
  const int lane = tid & 31;
  const int wave = tid >> 5;
  const int m16  = lane & 15;
  const int hh   = lane >> 4;
  const int wm   = wave >> 1;   // 0..3 (M dir)
  const int wn   = wave & 1;    // 0..1 (N dir)
  const int m0   = blockIdx.y * GBM;
  const int n0   = blockIdx.x * GBN;

  // D# group1 constants for the A tile (data_size=2B, pad on,
  // pad_interval code 3 = 16 DWORDs, pad_amount code 3 = 4 DWORDs)
  const u32 g1w0 = (1u << 16) | (1u << 20) | (3u << 22) | (3u << 25);
  const u32 g1w1 = ((u32)K & 0xFFFFu) << 16;                       // tensor_dim0 lo
  const u32 g1w2 = ((u32)K >> 16) | (((u32)M & 0xFFFFu) << 16);    // dim0 hi | dim1 lo
  const u32 g1w3 = ((u32)M >> 16) | ((u32)GBK << 16);              // dim1 hi | tile0
  const u32 g1w4 = (u32)GBM;                                       // tile1 (tile2=0)
  const u32 g1w5 = (u32)K;                                         // dim0 stride lo
  const u32 lds_as[2] = { lds_byte_offset(&As[0][0]), lds_byte_offset(&As[1][0]) };

  // per-thread B staging coordinates (vector global reads, N-major LDS stores)
  const int kk = tid >> 3;          // 0..31
  const int nn = (tid & 7) << 4;    // 0,16,..,112
  const bf16* bsrc = B + (size_t)kk * N + (n0 + nn);

  v8f acc[2][4];
#pragma unroll
  for (int i = 0; i < 2; ++i)
#pragma unroll
    for (int j = 0; j < 4; ++j)
      acc[i][j] = (v8f){0.f, 0.f, 0.f, 0.f, 0.f, 0.f, 0.f, 0.f};

  // ---- prologue: tile 0 into buffer 0 ----
  if (wave == 0)
    tdm_load_2d(lds_as[0], A + (size_t)m0 * K, g1w0, g1w1, g1w2, g1w3, g1w4, g1w5);
  {
    v8bf b0 = *(const v8bf*)bsrc;
    v8bf b1 = *(const v8bf*)(bsrc + 8);
#pragma unroll
    for (int j = 0; j < 8; ++j) {
      Bs[0][(nn + j) * LDT + kk]     = b0[j];
      Bs[0][(nn + 8 + j) * LDT + kk] = b1[j];
    }
  }
  if (wave == 0) __builtin_amdgcn_s_wait_tensorcnt(0);
  __syncthreads();

  int cur = 0;
  for (int k0 = 0; k0 < K; k0 += GBK) {
    const int nxt = cur ^ 1;
    const bool more = (k0 + GBK) < K;

    // kick off next A tile DMA (runs concurrently with the WMMAs below)
    if (more && wave == 0)
      tdm_load_2d(lds_as[nxt], A + (size_t)m0 * K + (k0 + GBK),
                  g1w0, g1w1, g1w2, g1w3, g1w4, g1w5);
    // issue next B tile global loads early (latency hidden under WMMAs)
    v8bf nb0, nb1;
    if (more) {
      const bf16* src = bsrc + (size_t)(k0 + GBK) * N;
      nb0 = *(const v8bf*)src;
      nb1 = *(const v8bf*)(src + 8);
      if (k0 + 2 * GBK < K)
        __builtin_prefetch(bsrc + (size_t)(k0 + 2 * GBK) * N, 0, 1);
    }

    // ---- compute on `cur` ----
    v16bf af[2], bfr[4];
#pragma unroll
    for (int mi = 0; mi < 2; ++mi)
      af[mi] = load_frag(&As[cur][(wm * 32 + mi * 16 + m16) * LDT], hh);
#pragma unroll
    for (int ni = 0; ni < 4; ++ni)
      bfr[ni] = load_frag(&Bs[cur][(wn * 64 + ni * 16 + m16) * LDT], hh);
#pragma unroll
    for (int mi = 0; mi < 2; ++mi)
#pragma unroll
      for (int ni = 0; ni < 4; ++ni)
        acc[mi][ni] = WMMA_BF16(af[mi], bfr[ni], acc[mi][ni]);

    // ---- publish next B tile, then wait for next A DMA ----
    if (more) {
#pragma unroll
      for (int j = 0; j < 8; ++j) {
        Bs[nxt][(nn + j) * LDT + kk]     = nb0[j];
        Bs[nxt][(nn + 8 + j) * LDT + kk] = nb1[j];
      }
      if (wave == 0) __builtin_amdgcn_s_wait_tensorcnt(0);
    }
    __syncthreads();
    cur = nxt;
  }

  // Epilogue: C/D layout => element (M = i + 8*hh, N = m16) in VGPR i.
  // N is a compile-time constant -> i*N folds into store immediate offsets.
  float* cbase = C + (size_t)(m0 + wm * 32 + 8 * hh) * N + (n0 + wn * 64 + m16);
#pragma unroll
  for (int mi = 0; mi < 2; ++mi)
#pragma unroll
    for (int ni = 0; ni < 4; ++ni) {
      float* cp = cbase + (size_t)mi * 16 * N + ni * 16;
#pragma unroll
      for (int i = 0; i < 8; ++i) cp[(size_t)i * N] = acc[mi][ni][i];
    }
}

// ---------------------------------------------------------------------------
// Kernel 3: RoPE + head split + convert.
// qkv fp32 [B,T,3C] -> q,k bf16 [B*H, T, 64] (row-major), v bf16 transposed
// [B*H, 64, T] so PV B-fragments load contiguously along the key dimension.
// One thread per (b, h, t, j) with j = rotation pair index in [0,32).
// ---------------------------------------------------------------------------
__global__ void rope_split(const float* __restrict__ qkv,
                           const float* __restrict__ fcos,
                           const float* __restrict__ fsin,
                           bf16* __restrict__ qb, bf16* __restrict__ kb,
                           bf16* __restrict__ vt) {
  int idx = blockIdx.x * blockDim.x + threadIdx.x;  // [0, B*H*T*32)
  int j = idx & 31;
  int t = (idx >> 5) & (Tt - 1);
  int h = (idx >> 16) & (Hh - 1);
  int b = idx >> 20;

  const float* p = qkv + (size_t)(b * Tt + t) * (3 * Cc) + h * HD + 2 * j;
  float c = fcos[t * 32 + j];
  float s = fsin[t * 32 + j];
  float q0 = p[0],        q1 = p[1];
  float k0 = p[Cc],       k1 = p[Cc + 1];
  float v0 = p[2 * Cc],   v1 = p[2 * Cc + 1];

  int bh = b * Hh + h;
  size_t o = ((size_t)bh * Tt + t) * HD + 2 * j;
  qb[o]     = f2bf(q0 * c - q1 * s);
  qb[o + 1] = f2bf(q0 * s + q1 * c);
  kb[o]     = f2bf(k0 * c - k1 * s);
  kb[o + 1] = f2bf(k0 * s + k1 * c);

  size_t vo = ((size_t)bh * HD + 2 * j) * Tt + t;
  vt[vo]      = f2bf(v0);
  vt[vo + Tt] = f2bf(v1);
}

// ---------------------------------------------------------------------------
// Kernel 4: flash attention (causal, online softmax), WMMA bf16.
// grid.x = B*H*(T/64); 4 waves/block; each wave owns a 16-row Q tile.
// Per 32-key block: 4 WMMAs for S = Q K^T, fp32 softmax with shfl_xor row
// reductions, P relayout via wave-private LDS, 4 WMMAs for O += P V.
// ---------------------------------------------------------------------------
__global__ __launch_bounds__(128) void flash_attn_wmma(
    const bf16* __restrict__ qb, const bf16* __restrict__ kb,
    const bf16* __restrict__ vt, bf16* __restrict__ yb) {
  __shared__ __align__(16) bf16 Pl[4][16 * 32];

  const int bh   = blockIdx.x >> 5;   // T/64 = 32 q-blocks per (b,h)
  const int qblk = blockIdx.x & 31;
  const int wave = threadIdx.x >> 5;
  const int lane = threadIdx.x & 31;
  const int m16  = lane & 15;
  const int hh   = lane >> 4;
  const int qrow0 = qblk * 64 + wave * 16;

  // Q fragments: 16x64 = two 16x32 A-fragments, loaded straight from global
  const bf16* qrow = qb + ((size_t)bh * Tt + qrow0 + m16) * HD;
  v16bf qf0 = load_frag(qrow, hh);
  v16bf qf1 = load_frag(qrow + 32, hh);

  v8f acc[4];
#pragma unroll
  for (int d = 0; d < 4; ++d)
    acc[d] = (v8f){0.f, 0.f, 0.f, 0.f, 0.f, 0.f, 0.f, 0.f};
  float mrun[8], lrun[8];
#pragma unroll
  for (int i = 0; i < 8; ++i) { mrun[i] = -3.0e38f; lrun[i] = 0.f; }

  const float scale = 0.125f;  // 1/sqrt(64)
  const int qlast = qrow0 + 15;

  for (int kbk = 0; kbk <= qlast; kbk += 32) {
    v8f s0 = (v8f){0.f, 0.f, 0.f, 0.f, 0.f, 0.f, 0.f, 0.f};
    v8f s1 = s0;
    {
      const bf16* kr0 = kb + ((size_t)bh * Tt + kbk + m16) * HD;
      s0 = WMMA_BF16(qf0, load_frag(kr0, hh), s0);
      s0 = WMMA_BF16(qf1, load_frag(kr0 + 32, hh), s0);
      const bf16* kr1 = kr0 + 16 * HD;
      s1 = WMMA_BF16(qf0, load_frag(kr1, hh), s1);
      s1 = WMMA_BF16(qf1, load_frag(kr1 + 32, hh), s1);
    }
    // scale + causal mask (element (M = i+8*hh, N = m16))
#pragma unroll
    for (int i = 0; i < 8; ++i) {
      int qg = qrow0 + i + 8 * hh;
      s0[i] = ((kbk + m16)      <= qg) ? s0[i] * scale : -3.0e38f;
      s1[i] = ((kbk + 16 + m16) <= qg) ? s1[i] * scale : -3.0e38f;
    }
    // online softmax; row stats live per-VGPR (row = i + 8*hh), reductions
    // across the 16 lanes of a half (xor masks < 16 never cross halves)
    float corr[8];
#pragma unroll
    for (int i = 0; i < 8; ++i) {
      float r = fmaxf(s0[i], s1[i]);
      r = fmaxf(r, __shfl_xor(r, 1));
      r = fmaxf(r, __shfl_xor(r, 2));
      r = fmaxf(r, __shfl_xor(r, 4));
      r = fmaxf(r, __shfl_xor(r, 8));
      float mnew = fmaxf(mrun[i], r);
      float p0 = __expf(s0[i] - mnew);
      float p1 = __expf(s1[i] - mnew);
      s0[i] = p0; s1[i] = p1;
      float rs = p0 + p1;
      rs += __shfl_xor(rs, 1);
      rs += __shfl_xor(rs, 2);
      rs += __shfl_xor(rs, 4);
      rs += __shfl_xor(rs, 8);
      float cf = __expf(mrun[i] - mnew);
      corr[i] = cf;
      lrun[i] = lrun[i] * cf + rs;
      mrun[i] = mnew;
    }
#pragma unroll
    for (int d = 0; d < 4; ++d)
#pragma unroll
      for (int i = 0; i < 8; ++i) acc[d][i] *= corr[i];

    // P (16x32 fp32, C/D layout) -> LDS bf16 -> reload as A-fragment.
    // LDS ops are in-order within a wave; the wait is a compiler/HW fence.
    bf16* pl = &Pl[wave][0];
#pragma unroll
    for (int i = 0; i < 8; ++i) {
      int prow = i + 8 * hh;
      pl[prow * 32 + m16]      = f2bf(s0[i]);
      pl[prow * 32 + 16 + m16] = f2bf(s1[i]);
    }
    asm volatile("s_wait_dscnt 0x0" ::: "memory");
    v16bf pf = load_frag(&pl[m16 * 32], hh);

    // O += P * V : V^T stored [64, T] so B-fragment chunks are contiguous in k
#pragma unroll
    for (int d = 0; d < 4; ++d) {
      const bf16* vcol = vt + ((size_t)bh * HD + d * 16 + m16) * Tt + kbk;
      acc[d] = WMMA_BF16(pf, load_frag(vcol, hh), acc[d]);
    }
  }

  // epilogue: normalize and scatter heads back to [B,T,C] bf16
  const int b = bh >> 4, h = bh & 15;
  float inv[8];
#pragma unroll
  for (int i = 0; i < 8; ++i) inv[i] = 1.0f / lrun[i];
#pragma unroll
  for (int d = 0; d < 4; ++d)
#pragma unroll
    for (int i = 0; i < 8; ++i) {
      int row = qrow0 + i + 8 * hh;
      int col = d * 16 + m16;
      yb[((size_t)(b * Tt + row)) * Cc + h * HD + col] = f2bf(acc[d][i] * inv[i]);
    }
}

// ---------------------------------------------------------------------------
// Host-side launcher
// ---------------------------------------------------------------------------
extern "C" void kernel_launch(void* const* d_in, const int* in_sizes, int n_in,
                              void* d_out, int out_size, void* d_ws, size_t ws_size,
                              hipStream_t stream) {
  (void)in_sizes; (void)n_in; (void)out_size; (void)ws_size;

  const float* x      = (const float*)d_in[0];  // [2,2048,1024]
  const float* w_qkv  = (const float*)d_in[1];  // [1024,3072]
  const float* w_proj = (const float*)d_in[2];  // [1024,1024]
  const float* fcos   = (const float*)d_in[3];  // [2048,32]
  const float* fsin   = (const float*)d_in[4];  // [2048,32]
  float* out = (float*)d_out;                   // [2,2048,1024]

  // workspace carve-up (~96 MB total)
  char* w = (char*)d_ws;
  bf16* x_bf     = (bf16*)w;  w += (size_t)4096 * 1024 * 2;
  bf16* wqkv_bf  = (bf16*)w;  w += (size_t)1024 * 3072 * 2;
  bf16* wproj_bf = (bf16*)w;  w += (size_t)1024 * 1024 * 2;
  float* qkv     = (float*)w; w += (size_t)4096 * 3072 * 4;
  bf16* qb       = (bf16*)w;  w += (size_t)32 * 2048 * 64 * 2;
  bf16* kbuf     = (bf16*)w;  w += (size_t)32 * 2048 * 64 * 2;
  bf16* vtb      = (bf16*)w;  w += (size_t)32 * 2048 * 64 * 2;
  bf16* yb       = (bf16*)w;

  // 1) convert inputs to bf16 (4 elements / thread)
  cvt_f32_to_bf16<<<4194304 / 4 / 256, 256, 0, stream>>>(x, x_bf, 4194304 / 4);
  cvt_f32_to_bf16<<<3145728 / 4 / 256, 256, 0, stream>>>(w_qkv, wqkv_bf, 3145728 / 4);
  cvt_f32_to_bf16<<<1048576 / 4 / 256, 256, 0, stream>>>(w_proj, wproj_bf, 1048576 / 4);

  // 2) qkv = x @ w_qkv  : [4096,1024] x [1024,3072]
  gemm_bf16_wmma<4096, 3072, 1024><<<dim3(3072 / GBN, 4096 / GBM), 256, 0, stream>>>(
      x_bf, wqkv_bf, qkv);

  // 3) RoPE + head split + V transpose
  rope_split<<<2097152 / 256, 256, 0, stream>>>(qkv, fcos, fsin, qb, kbuf, vtb);

  // 4) causal flash attention
  flash_attn_wmma<<<Bb * Hh * (Tt / 64), 128, 0, stream>>>(qb, kbuf, vtb, yb);

  // 5) out = y @ w_proj : [4096,1024] x [1024,1024]
  gemm_bf16_wmma<4096, 1024, 1024><<<dim3(1024 / GBN, 4096 / GBM), 256, 0, stream>>>(
      yb, wproj_bf, out);
}
